// Block_70325794504976
// MI455X (gfx1250) — matrix operand order repaired and checked by reading, not compiled, not run on hardware
//
#include <hip/hip_runtime.h>

typedef __attribute__((ext_vector_type(16))) _Float16 v16h;
typedef __attribute__((ext_vector_type(8)))  _Float16 v8h;
typedef __attribute__((ext_vector_type(4)))  _Float16 v4h;
typedef __attribute__((ext_vector_type(8)))  float    v8f;

namespace {
constexpr int kB   = 16;     // batch
constexpr int kC   = 1024;   // channels
constexpr int kS   = 1024;   // spatial (32*32)
constexpr int kCTX = 2048;   // context dim
constexpr int kSK  = 256;    // context length
constexpr int kH   = 16;     // heads
constexpr int kD   = 64;     // head dim
}

__device__ __forceinline__ int lane_id() { return (int)(threadIdx.x & 31); }

__device__ __forceinline__ v8f wmma16(v16h a, v16h b, v8f c) {
  return __builtin_amdgcn_wmma_f32_16x16x32_f16(false, a, false, b, (short)0, c,
                                                false, false);
}

// CDNA5 async global->LDS copy, 128b per lane, ASYNCcnt-tracked.
__device__ __forceinline__ void async_copy_b128(unsigned lds_off,
                                                const void* gaddr) {
  asm volatile("global_load_async_to_lds_b128 %0, %1, off"
               :: "v"(lds_off), "v"(gaddr) : "memory");
}
__device__ __forceinline__ void async_wait0() {
  asm volatile("s_wait_asynccnt 0x0" ::: "memory");
}

// A-fragment (16x32 f16), element (m,k) at p[m*ms + k], k contiguous.
// Lane layout: m = lane&15; h = lane>>4 supplies K = h*16+j -> two 16B LDS
// loads.
__device__ __forceinline__ v16h load_A_frag_c(const _Float16* p, int ms) {
  const int l = lane_id();
  const int h = l >> 4, m = l & 15;
  const v8h lo = *(const v8h*)(p + m * ms + h * 16);
  const v8h hi = *(const v8h*)(p + m * ms + h * 16 + 8);
  v16h r;
#pragma unroll
  for (int j = 0; j < 8; ++j) { r[j] = lo[j]; r[8 + j] = hi[j]; }
  return r;
}

// B-fragment (32x16 f16), element (k,n) at p[n*ns + k], k contiguous.
// Lane layout: n = lane&15; h = lane>>4 supplies K = h*16+j -> two 16B loads.
__device__ __forceinline__ v16h load_B_frag_c(const _Float16* p, int ns) {
  const int l = lane_id();
  const int h = l >> 4, n = l & 15;
  const v8h lo = *(const v8h*)(p + n * ns + h * 16);
  const v8h hi = *(const v8h*)(p + n * ns + h * 16 + 8);
  v16h r;
#pragma unroll
  for (int j = 0; j < 8; ++j) { r[j] = lo[j]; r[8 + j] = hi[j]; }
  return r;
}

// B fragment for P (attention probs): B[k=s][n=t] = exp(logit[t][s] - max[t])
__device__ __forceinline__ v16h load_P_frag(const _Float16* Ls, int ldl,
                                            const float* Mx, int s0, int t0l) {
  const int l = lane_id();
  const int h = l >> 4, n = l & 15;
  const int t = t0l + n;
  const float m = Mx[t];
  v16h r;
#pragma unroll
  for (int j = 0; j < 16; ++j) {
    const float lv = (float)Ls[t * ldl + (s0 + h * 16 + j)];
    r[j] = (_Float16)__expf(lv - m);
  }
  return r;
}

// ---------------------------------------------------------------------------
// GroupNorm over x[b, c, s]: 32 groups of 32 consecutive channels (contiguous
// 32768-float span per (b,g)).  Output f16.
// ---------------------------------------------------------------------------
__global__ __launch_bounds__(256) void gn_x_kernel(
    const float* __restrict__ x, const float* __restrict__ gamma,
    const float* __restrict__ beta, _Float16* __restrict__ xn) {
  constexpr int CPG = kC / 32;
  constexpr int NE  = CPG * kS;
  const int b = blockIdx.x >> 5, g = blockIdx.x & 31;
  const float* xp = x + ((long)b * kC + (long)g * CPG) * kS;
  _Float16* xo = xn + ((long)b * kC + (long)g * CPG) * kS;
  const int tid = threadIdx.x;

  float s = 0.f, ss = 0.f;
  for (int i = tid; i < NE; i += 256) {
    const float v = xp[i];
    s += v; ss += v * v;
  }
  __shared__ float r0[256], r1[256];
  r0[tid] = s; r1[tid] = ss;
  __syncthreads();
  for (int st = 128; st > 0; st >>= 1) {
    if (tid < st) { r0[tid] += r0[tid + st]; r1[tid] += r1[tid + st]; }
    __syncthreads();
  }
  const float mean = r0[0] * (1.0f / NE);
  const float var  = r1[0] * (1.0f / NE) - mean * mean;
  const float inv  = rsqrtf(var + 1e-5f);
  for (int i = tid; i < NE; i += 256) {
    const int ch = g * CPG + (i >> 10);
    const float v = (xp[i] - mean) * inv * gamma[ch] + beta[ch];
    xo[i] = (_Float16)v;
  }
}

// ---------------------------------------------------------------------------
// GroupNorm over transposed context: ctx[b, ch, s] = context[b, s, ch].
// ---------------------------------------------------------------------------
__global__ __launch_bounds__(256) void gn_ctx_kernel(
    const float* __restrict__ context, const float* __restrict__ gamma,
    const float* __restrict__ beta, _Float16* __restrict__ cn) {
  constexpr int CPG = kCTX / 32;
  constexpr int NE  = CPG * kSK;
  const int b = blockIdx.x >> 5, g = blockIdx.x & 31;
  const int ch0 = g * CPG;
  const float* cb = context + (long)b * kSK * kCTX;
  const int tid = threadIdx.x;

  float s = 0.f, ss = 0.f;
  for (int i = tid; i < NE; i += 256) {
    const int ch = ch0 + (i >> 8);
    const int sp = i & (kSK - 1);
    const float v = cb[(long)sp * kCTX + ch];
    s += v; ss += v * v;
  }
  __shared__ float r0[256], r1[256];
  r0[tid] = s; r1[tid] = ss;
  __syncthreads();
  for (int st = 128; st > 0; st >>= 1) {
    if (tid < st) { r0[tid] += r0[tid + st]; r1[tid] += r1[tid + st]; }
    __syncthreads();
  }
  const float mean = r0[0] * (1.0f / NE);
  const float var  = r1[0] * (1.0f / NE) - mean * mean;
  const float inv  = rsqrtf(var + 1e-5f);
  for (int i = tid; i < NE; i += 256) {
    const int ch = ch0 + (i >> 8);
    const int sp = i & (kSK - 1);
    float v = cb[(long)sp * kCTX + ch];
    v = (v - mean) * inv * gamma[ch] + beta[ch];
    cn[((long)b * kCTX + ch) * kSK + sp] = (_Float16)v;
  }
}

// ---------------------------------------------------------------------------
// Batched GEMM: O = scale * (A @ Bmat[b] + bias), A fp32 [M,K], Bmat f16 [K,N].
// MODE 0: f16 out, row-major [m][n] (stride ohStride per batch)
// MODE 1: f16 out, head-major [b][m>>6][n][m&63]   (for Q / K layouts)
// MODE 2: f32 out + residual
// Tiles 128x128x64; 8 waves; per wave 2x4 WMMA tiles; B staged transposed so
// both fragment kinds are two ds_load_b128 each.
// ---------------------------------------------------------------------------
template <int MODE>
__global__ __launch_bounds__(256) void gemm_kernel(
    const float* __restrict__ A, const float* __restrict__ bias,
    const _Float16* __restrict__ Bmat, long bStride,
    _Float16* __restrict__ Oh, long ohStride,
    float* __restrict__ Of, const float* __restrict__ Rsd, long ofStride,
    float scale, int N, int K) {
  constexpr int BM = 128, BN = 128, BK = 64;
  __shared__ _Float16 As[BM][BK + 8];    // [m][k], k contiguous
  __shared__ _Float16 BsT[BN][BK + 8];   // [n][k], k contiguous (transposed)

  const int batch = blockIdx.z;
  const int m0 = blockIdx.y * BM;
  const int n0 = blockIdx.x * BN;
  const _Float16* Bp = Bmat + (long)batch * bStride;

  const int tid = threadIdx.x;
  const int lane = tid & 31;
  const int wid = tid >> 5;
  const int wm = wid & 3;
  const int wn = wid >> 2;

  const v8f vz = {0.f, 0.f, 0.f, 0.f, 0.f, 0.f, 0.f, 0.f};
  v8f acc[2][4];
#pragma unroll
  for (int i = 0; i < 2; ++i)
#pragma unroll
    for (int j = 0; j < 4; ++j) acc[i][j] = vz;

  for (int k0 = 0; k0 < K; k0 += BK) {
    {  // stage A tile fp32->f16 (packed 8B stores)
      const int col = (tid & 15) * 4;
      const int rb  = tid >> 4;
#pragma unroll
      for (int r = 0; r < 8; ++r) {
        const int row = rb + r * 16;
        const float4 f = *(const float4*)(A + (long)(m0 + row) * K + k0 + col);
        const v4h hv = {(_Float16)f.x, (_Float16)f.y, (_Float16)f.z,
                        (_Float16)f.w};
        *(v4h*)&As[row][col] = hv;
      }
    }
    {  // stage B tile transposed: read [k][n] rows vectorized, scatter to [n][k]
      const int nc = (tid & 15) * 8;
      const int kb_ = tid >> 4;
#pragma unroll
      for (int r = 0; r < 4; ++r) {
        const int krow = kb_ + r * 16;
        const v8h hv = *(const v8h*)(Bp + (long)(k0 + krow) * N + n0 + nc);
#pragma unroll
        for (int e = 0; e < 8; ++e) BsT[nc + e][krow] = hv[e];
      }
    }
    if (k0 + BK < K) {  // pull next tiles toward L2 (global_prefetch_b8)
      __builtin_prefetch(A + (long)(m0 + (tid >> 1)) * K + k0 + BK, 0, 0);
      __builtin_prefetch(Bp + (long)(k0 + BK + (tid >> 2)) * N + n0, 0, 0);
    }
    __syncthreads();
#pragma unroll
    for (int kk = 0; kk < 2; ++kk) {
      const v16h a0 = load_A_frag_c(&As[wm * 32 +  0][kk * 32], BK + 8);
      const v16h a1 = load_A_frag_c(&As[wm * 32 + 16][kk * 32], BK + 8);
#pragma unroll
      for (int j = 0; j < 4; ++j) {
        const v16h bf = load_B_frag_c(&BsT[wn * 64 + j * 16][kk * 32], BK + 8);
        acc[0][j] = wmma16(a0, bf, acc[0][j]);
        acc[1][j] = wmma16(a1, bf, acc[1][j]);
      }
    }
    __syncthreads();
  }

  // epilogue: C/D layout -> m = r + 8*(lane>=16), n = lane&15
  const int h = lane >> 4, nl = lane & 15;
#pragma unroll
  for (int i = 0; i < 2; ++i) {
#pragma unroll
    for (int j = 0; j < 4; ++j) {
      const int nb = n0 + wn * 64 + j * 16 + nl;
#pragma unroll
      for (int r = 0; r < 8; ++r) {
        const int gm = m0 + wm * 32 + i * 16 + r + 8 * h;
        const float vv = (acc[i][j][r] + bias[gm]) * scale;
        if constexpr (MODE == 2) {
          const long idx = (long)batch * ofStride + (long)gm * N + nb;
          Of[idx] = Rsd[idx] + vv;
        } else if constexpr (MODE == 1) {
          const long idx =
              (((long)batch * kH + (gm >> 6)) * (long)N + nb) * kD + (gm & 63);
          Oh[idx] = (_Float16)vv;
        } else {
          Oh[(long)batch * ohStride + (long)gm * N + nb] = (_Float16)vv;
        }
      }
    }
  }
}

// ---------------------------------------------------------------------------
// Fused masked attention for one (batch, head, 64-row q block).
// q: [b][h][t][d], k: [b][h][s][d], v: [b][h][d][s]; scale folded into q,k.
// Q/K/V tiles staged via async global->LDS b128; fragments are 2x ds_load_b128.
// ---------------------------------------------------------------------------
__global__ __launch_bounds__(256) void attn_kernel(
    const _Float16* __restrict__ q, const _Float16* __restrict__ k,
    const _Float16* __restrict__ v, const int* __restrict__ mask,
    _Float16* __restrict__ a) {
  __shared__ _Float16 Qs[64][kD];        // [t][d]   8 KB
  __shared__ _Float16 Ks[kSK][kD];       // [s][d]  32 KB
  __shared__ _Float16 Vs[kD][kSK];       // [d][s]  32 KB
  __shared__ _Float16 Ls[64][kSK + 8];   // logits  33 KB
  __shared__ float red0[256], red1[256];
  __shared__ float Mx[64], Zs[64];
  __shared__ int msk[kSK];

  const int b = blockIdx.z, hh = blockIdx.y, tb = blockIdx.x;
  const int t0 = tb * 64;
  const int tid = threadIdx.x, lane = tid & 31, wid = tid >> 5;
  const v8f vz = {0.f, 0.f, 0.f, 0.f, 0.f, 0.f, 0.f, 0.f};

  // ---- async stage Q (8 KB), K (32 KB), V (32 KB) ----
  {
    const _Float16* qg = q + (((long)b * kH + hh) * kS + t0) * kD;
    const _Float16* kg = k + ((long)b * kH + hh) * (long)(kSK * kD);
    const _Float16* vg = v + ((long)b * kH + hh) * (long)(kD * kSK);
    const unsigned qo = (unsigned)(size_t)&Qs[0][0];
    const unsigned ko = (unsigned)(size_t)&Ks[0][0];
    const unsigned vo = (unsigned)(size_t)&Vs[0][0];
#pragma unroll
    for (int i = 0; i < 2; ++i)
      async_copy_b128(qo + (tid + i * 256) * 16,
                      (const char*)qg + (tid + i * 256) * 16);
#pragma unroll
    for (int i = 0; i < 8; ++i)
      async_copy_b128(ko + (tid + i * 256) * 16,
                      (const char*)kg + (tid + i * 256) * 16);
#pragma unroll
    for (int i = 0; i < 8; ++i)
      async_copy_b128(vo + (tid + i * 256) * 16,
                      (const char*)vg + (tid + i * 256) * 16);
  }
  if (tid < kSK) msk[tid] = mask[b * kSK + tid];
  async_wait0();
  __syncthreads();

  // ---- logits = Q K^T (wave: one t-tile x 8 s-tiles, K=64 in 2 steps) ----
  {
    const int t_tile = wid & 3;
    const int s_half = wid >> 2;
    const v16h aq0 = load_A_frag_c(&Qs[t_tile * 16][0],  kD);  // d 0..31
    const v16h aq1 = load_A_frag_c(&Qs[t_tile * 16][32], kD);  // d 32..63
    v8f accq[8];
#pragma unroll
    for (int j = 0; j < 8; ++j) accq[j] = vz;
#pragma unroll
    for (int j = 0; j < 8; ++j) {
      const int s0 = s_half * 128 + j * 16;
      const v16h b0 = load_B_frag_c(&Ks[s0][0],  kD);
      const v16h b1 = load_B_frag_c(&Ks[s0][32], kD);
      accq[j] = wmma16(aq0, b0, accq[j]);
      accq[j] = wmma16(aq1, b1, accq[j]);
    }
    const int h = lane >> 4, nl = lane & 15;
#pragma unroll
    for (int j = 0; j < 8; ++j) {
      const int sc = s_half * 128 + j * 16 + nl;
      const bool mk = msk[sc] != 0;
#pragma unroll
      for (int r = 0; r < 8; ++r) {
        const int tt = t_tile * 16 + r + 8 * h;
        Ls[tt][sc] = (_Float16)(mk ? accq[j][r] : -10000.0f);
      }
    }
  }
  __syncthreads();

  // ---- softmax stats (4 threads per row); 1/Z deferred to epilogue ----
  {
    const int row = tid >> 2, part = tid & 3;
    float mx = -3.0e4f;
    for (int s = part * 64; s < part * 64 + 64; ++s)
      mx = fmaxf(mx, (float)Ls[row][s]);
    red0[tid] = mx;
    __syncthreads();
    const float m4 = fmaxf(fmaxf(red0[row * 4 + 0], red0[row * 4 + 1]),
                           fmaxf(red0[row * 4 + 2], red0[row * 4 + 3]));
    float sum = 0.f;
    for (int s = part * 64; s < part * 64 + 64; ++s)
      sum += __expf((float)Ls[row][s] - m4);
    red1[tid] = sum;
    __syncthreads();
    if (part == 0) {
      Mx[row] = m4;
      Zs[row] = red1[row * 4 + 0] + red1[row * 4 + 1] +
                red1[row * 4 + 2] + red1[row * 4 + 3];
    }
  }
  __syncthreads();

  // ---- out[d, t] = (P V) / Z : wave handles 2 d-tiles x 1 t-tile ----
  {
    const int t_tile = wid & 3;
    const int d_half = wid >> 2;
    v8f accp[2];
    accp[0] = vz; accp[1] = vz;
#pragma unroll
    for (int ks = 0; ks < 8; ++ks) {
      const int s0 = ks * 32;
      const v16h pb = load_P_frag(&Ls[0][0], kSK + 8, Mx, s0, t_tile * 16);
      const v16h a0 = load_A_frag_c(&Vs[d_half * 32 +  0][s0], kSK);
      const v16h a1 = load_A_frag_c(&Vs[d_half * 32 + 16][s0], kSK);
      accp[0] = wmma16(a0, pb, accp[0]);
      accp[1] = wmma16(a1, pb, accp[1]);
    }
    const int h = lane >> 4, nl = lane & 15;
    _Float16* ab = a + ((long)b * kC + (long)hh * kD) * kS;  // [c][t]
    const int tt = t0 + t_tile * 16 + nl;
    const float z = 1.0f / Zs[t_tile * 16 + nl];
#pragma unroll
    for (int i = 0; i < 2; ++i) {
#pragma unroll
      for (int r = 0; r < 8; ++r) {
        const int dd = d_half * 32 + i * 16 + r + 8 * h;
        ab[(long)dd * kS + tt] = (_Float16)(accp[i][r] * z);
      }
    }
  }
}

// ---------------------------------------------------------------------------
extern "C" void kernel_launch(void* const* d_in, const int* in_sizes, int n_in,
                              void* d_out, int out_size, void* d_ws,
                              size_t ws_size, hipStream_t stream) {
  const float* x       = (const float*)d_in[0];
  const float* context = (const float*)d_in[1];
  const int*   mask    = (const int*)d_in[2];
  const float* gamma_x = (const float*)d_in[3];
  const float* beta_x  = (const float*)d_in[4];
  const float* gamma_c = (const float*)d_in[5];
  const float* beta_c  = (const float*)d_in[6];
  const float* Wq      = (const float*)d_in[7];
  const float* bq      = (const float*)d_in[8];
  const float* Wkv     = (const float*)d_in[9];
  const float* bkv     = (const float*)d_in[10];
  const float* Wp      = (const float*)d_in[11];
  const float* bp      = (const float*)d_in[12];
  float* out = (float*)d_out;

  _Float16* xn   = (_Float16*)d_ws;                    // [B][C][S]
  _Float16* ctxn = xn   + (long)kB * kC * kS;          // [B][CTX][SK]
  _Float16* qb   = ctxn + (long)kB * kCTX * kSK;       // [B][H][S][D] (scaled)
  _Float16* kb   = qb   + (long)kB * kC * kS;          // [B][H][SK][D] (scaled)
  _Float16* vb   = kb   + (long)kB * kC * kSK;         // [B][H][D][SK]
  _Float16* ab   = vb   + (long)kB * kC * kSK;         // [B][C][S]

  const float scale = 0.35355339059327379f;  // 64^-0.25

  gn_x_kernel<<<dim3(kB * 32), 256, 0, stream>>>(x, gamma_x, beta_x, xn);
  gn_ctx_kernel<<<dim3(kB * 32), 256, 0, stream>>>(context, gamma_c, beta_c,
                                                   ctxn);

  // q = scale * (Wq @ xn + bq), head-major layout
  gemm_kernel<1><<<dim3(kS / 128, kC / 128, kB), 256, 0, stream>>>(
      Wq, bq, xn, (long)kC * kS, qb, 0,
      nullptr, nullptr, 0, scale, kS, kC);
  // k = scale * (Wkv[:C] @ ctxn + bkv[:C]), head-major layout
  gemm_kernel<1><<<dim3(kSK / 128, kC / 128, kB), 256, 0, stream>>>(
      Wkv, bkv, ctxn, (long)kCTX * kSK, kb, 0,
      nullptr, nullptr, 0, scale, kSK, kCTX);
  // v = Wkv[C:] @ ctxn + bkv[C:], channel-major ([b][h][d][s])
  gemm_kernel<0><<<dim3(kSK / 128, kC / 128, kB), 256, 0, stream>>>(
      Wkv + (long)kC * kCTX, bkv + kC, ctxn, (long)kCTX * kSK, vb,
      (long)kC * kSK, nullptr, nullptr, 0, 1.0f, kSK, kCTX);

  attn_kernel<<<dim3(kS / 64, kH, kB), 256, 0, stream>>>(qb, kb, vb, mask, ab);

  // out = x + (Wp @ a + bp)
  gemm_kernel<2><<<dim3(kS / 128, kC / 128, kB), 256, 0, stream>>>(
      Wp, bp, ab, (long)kC * kS, nullptr, 0, out, x, (long)kC * kS,
      1.0f, kS, kC);
}
